// TransformerBlock_85822036509373
// MI455X (gfx1250) — matrix operand order repaired and checked by reading, not compiled, not run on hardware
//
#include <hip/hip_runtime.h>
#include <cmath>

#define N_EMBD 1024
#define N_HEAD 16
#define HEAD_DIM 64
#define SEQ 2048
#define BATCH 4
#define NTOK (BATCH * SEQ) /* 8192 */

typedef __bf16 bf16;
typedef __attribute__((ext_vector_type(16))) __bf16 v16bf;
typedef __attribute__((ext_vector_type(8)))  __bf16 v8bf;
typedef __attribute__((ext_vector_type(8)))  float  v8f;
typedef __attribute__((__vector_size__(4 * sizeof(int)))) int v4i;

#ifndef __has_builtin
#define __has_builtin(x) 0
#endif

#if __has_builtin(__builtin_amdgcn_global_load_async_to_lds_b128)
#define HAVE_ASYNC_LDS 1
#else
#define HAVE_ASYNC_LDS 0
#endif

#if HAVE_ASYNC_LDS
typedef __attribute__((address_space(1))) v4i* gas_v4i_ptr;
typedef __attribute__((address_space(3))) v4i* las_v4i_ptr;
#endif

// 16-byte global -> LDS copy; async (ASYNCcnt) when available.
__device__ __forceinline__ void cp16(bf16* dst_lds, const bf16* src_g) {
#if HAVE_ASYNC_LDS
  __builtin_amdgcn_global_load_async_to_lds_b128(
      (gas_v4i_ptr)(src_g), (las_v4i_ptr)(dst_lds), 0, 0);
#else
  *(v8bf*)dst_lds = *(const v8bf*)src_g;
#endif
}

__device__ __forceinline__ void wait_async() {
#if HAVE_ASYNC_LDS
#if __has_builtin(__builtin_amdgcn_s_wait_asynccnt)
  __builtin_amdgcn_s_wait_asynccnt(0);
#else
  asm volatile("s_wait_asynccnt 0x0" ::: "memory");
#endif
#endif
}

__device__ __forceinline__ v8f wmma_bf16f32(v16bf a, v16bf b, v8f c) {
  // (neg_a, A, neg_b, B, c_mod, C, reuse_a, reuse_b)
  return __builtin_amdgcn_wmma_f32_16x16x32_bf16(false, a, false, b, (short)0, c,
                                                 false, false);
}

__device__ __forceinline__ v16bf pack16(v8bf lo, v8bf hi) {
  v16bf r;
#pragma unroll
  for (int i = 0; i < 8; ++i) { r[i] = lo[i]; r[i + 8] = hi[i]; }
  return r;
}

__device__ __forceinline__ v8f zero8() {
  v8f z;
#pragma unroll
  for (int i = 0; i < 8; ++i) z[i] = 0.0f;
  return z;
}

// ---------------------------------------------------------------------------
// Coalesced fp32 -> bf16 transposed weight convert: wt[n*K + k] = w[k*N + n]
// 32x32 tile through LDS; block = 256 threads (32x8).
// ---------------------------------------------------------------------------
__global__ __launch_bounds__(256) void transpose_cvt_kernel(
    const float* __restrict__ w, bf16* __restrict__ wt, int K, int N) {
  __shared__ float tile[32][33];
  const int n0 = blockIdx.x * 32, k0 = blockIdx.y * 32;
  const int tx = threadIdx.x & 31, ty = threadIdx.x >> 5;
#pragma unroll
  for (int i = 0; i < 32; i += 8)
    tile[ty + i][tx] = w[(size_t)(k0 + ty + i) * N + n0 + tx];
  __syncthreads();
#pragma unroll
  for (int i = 0; i < 32; i += 8)
    wt[(size_t)(n0 + ty + i) * K + k0 + tx] = (bf16)tile[tx][ty + i];
}

// ---------------------------------------------------------------------------
// LayerNorm over C=1024, one row per 256-thread block, bf16 output
// ---------------------------------------------------------------------------
__global__ __launch_bounds__(256) void layernorm_kernel(
    const float* __restrict__ x, const float* __restrict__ w,
    const float* __restrict__ b, bf16* __restrict__ out) {
  const int row = blockIdx.x;
  const int tid = threadIdx.x, lane = tid & 31, wid = tid >> 5;
  const float4 v = *(const float4*)(x + (size_t)row * N_EMBD + tid * 4);
  float s = v.x + v.y + v.z + v.w;
  float sq = v.x * v.x + v.y * v.y + v.z * v.z + v.w * v.w;
#pragma unroll
  for (int off = 16; off > 0; off >>= 1) {
    s  += __shfl_xor(s, off, 32);
    sq += __shfl_xor(sq, off, 32);
  }
  __shared__ float ss[8], ssq[8];
  if (lane == 0) { ss[wid] = s; ssq[wid] = sq; }
  __syncthreads();
  float tot = 0.f, totq = 0.f;
#pragma unroll
  for (int i = 0; i < 8; ++i) { tot += ss[i]; totq += ssq[i]; }
  const float mean = tot * (1.0f / N_EMBD);
  const float var  = totq * (1.0f / N_EMBD) - mean * mean;
  const float rstd = rsqrtf(var + 1e-5f);
  const int c = tid * 4;
  const float xe[4] = {v.x, v.y, v.z, v.w};
#pragma unroll
  for (int i = 0; i < 4; ++i)
    out[(size_t)row * N_EMBD + c + i] =
        (bf16)(w[c + i] * (xe[i] - mean) * rstd + b[c + i]);
}

// ---------------------------------------------------------------------------
// GEMM: out[M,N] = epilogue(A[M,K] @ Wt[N,K]^T + bias)
//  MODE 0: bf16 out (bias)          MODE 1: f32 out = residual + acc + bias
//  MODE 2: bf16 out (bias + GELU)
// Block tile 128x128, BK=64 (2 WMMA K-steps), double-buffered async LDS.
// 8 waves, each 64x32 (4x2 WMMA tiles) -> 16 WMMAs per K-block per wave.
// ---------------------------------------------------------------------------
template <int MODE>
__global__ __launch_bounds__(256) void gemm_bf16_kernel(
    const bf16* __restrict__ A, const bf16* __restrict__ Wt,
    const float* __restrict__ bias, const float* __restrict__ residual,
    bf16* __restrict__ outB, float* __restrict__ outF, int M, int N, int K) {
  __shared__ __align__(16) bf16 sA[2][128][72];
  __shared__ __align__(16) bf16 sB[2][128][72];
  const int tid = threadIdx.x;
  const int lane = tid & 31, wid = tid >> 5;
  const int l16 = lane & 15, hlf = lane >> 4;
  const int wm = (wid & 1) * 64;   // wave row offset within block tile
  const int wn = (wid >> 1) * 32;  // wave col offset within block tile
  const int mBase = blockIdx.y * 128;
  const int nBase = blockIdx.x * 128;

  auto loadTiles = [&](int buf, int k0) {
#pragma unroll
    for (int t = 0; t < 4; ++t) {
      const int idx = tid + t * 256;          // 128 rows x 8 chunks of 8 bf16
      const int row = idx >> 3, c = (idx & 7) * 8;
      cp16(&sA[buf][row][c], A  + (size_t)(mBase + row) * K + k0 + c);
      cp16(&sB[buf][row][c], Wt + (size_t)(nBase + row) * K + k0 + c);
    }
  };

  v8f acc[4][2];
#pragma unroll
  for (int mt = 0; mt < 4; ++mt)
#pragma unroll
    for (int nt = 0; nt < 2; ++nt) acc[mt][nt] = zero8();

  loadTiles(0, 0);
  wait_async();
  __syncthreads();

  int buf = 0;
  for (int k0 = 0; k0 < K; k0 += 64) {
    if (k0 + 64 < K) loadTiles(buf ^ 1, k0 + 64);  // overlap with compute

#pragma unroll
    for (int ks = 0; ks < 2; ++ks) {
      v16bf af[4], bfr[2];
#pragma unroll
      for (int mt = 0; mt < 4; ++mt) {
        const int r = wm + mt * 16 + l16;
        af[mt] = pack16(*(const v8bf*)&sA[buf][r][ks * 32 + hlf * 8],
                        *(const v8bf*)&sA[buf][r][ks * 32 + 16 + hlf * 8]);
      }
#pragma unroll
      for (int nt = 0; nt < 2; ++nt) {
        const int r = wn + nt * 16 + l16;
        bfr[nt] = pack16(*(const v8bf*)&sB[buf][r][ks * 32 + hlf * 8],
                         *(const v8bf*)&sB[buf][r][ks * 32 + 16 + hlf * 8]);
      }
#pragma unroll
      for (int mt = 0; mt < 4; ++mt)
#pragma unroll
        for (int nt = 0; nt < 2; ++nt)
          acc[mt][nt] = wmma_bf16f32(af[mt], bfr[nt], acc[mt][nt]);
    }

    wait_async();
    __syncthreads();
    buf ^= 1;
  }

#pragma unroll
  for (int mt = 0; mt < 4; ++mt) {
#pragma unroll
    for (int nt = 0; nt < 2; ++nt) {
      const int col = nBase + wn + nt * 16 + l16;
      const float bv = bias[col];
#pragma unroll
      for (int r = 0; r < 8; ++r) {
        const int row = mBase + wm + mt * 16 + r + hlf * 8;
        float v = acc[mt][nt][r] + bv;
        if (MODE == 2) {
          const float u = 0.7978845608028654f * (v + 0.044715f * v * v * v);
          v = 0.5f * v * (1.0f + tanhf(u));
        }
        if (MODE == 1)
          outF[(size_t)row * N + col] = residual[(size_t)row * N + col] + v;
        else
          outB[(size_t)row * N + col] = (bf16)v;
      }
    }
  }
}

// ---------------------------------------------------------------------------
// Flash attention (causal): block = (qtile of 128 rows, head, batch).
// Each of 8 waves owns a 16-row q strip. S = Q K^T and O += P V via WMMA.
// ---------------------------------------------------------------------------
__global__ __launch_bounds__(256) void attn_kernel(const bf16* __restrict__ qkv,
                                                   bf16* __restrict__ y) {
  const int C3 = 3 * N_EMBD;
  const int qt = blockIdx.x, h = blockIdx.y, b = blockIdx.z;
  const int qbase = qt * 128;
  const int tid = threadIdx.x, lane = tid & 31, wid = tid >> 5;
  const int l16 = lane & 15, hlf = lane >> 4;

  __shared__ __align__(16) bf16 sQ[128][72];
  __shared__ __align__(16) bf16 sK[64][72];
  __shared__ __align__(16) bf16 sVt[64][72];   // transposed: [d][s]
  __shared__ __align__(16) bf16 sP[128][72];

  const bf16* qp = qkv + (size_t)(b * SEQ + qbase) * C3 + h * HEAD_DIM;
#pragma unroll
  for (int t = 0; t < 4; ++t) {
    const int idx = tid + t * 256;             // 128 rows x 8 chunks
    const int row = idx >> 3, c = (idx & 7) * 8;
    cp16(&sQ[row][c], qp + (size_t)row * C3 + c);
  }
  wait_async();
  __syncthreads();

  v16bf qf[2];
  {
    const int r = wid * 16 + l16;
#pragma unroll
    for (int ks = 0; ks < 2; ++ks)
      qf[ks] = pack16(*(const v8bf*)&sQ[r][ks * 32 + hlf * 8],
                      *(const v8bf*)&sQ[r][ks * 32 + 16 + hlf * 8]);
  }

  v8f o[4];
#pragma unroll
  for (int dt = 0; dt < 4; ++dt) o[dt] = zero8();
  float mrow[8], lrow[8];
#pragma unroll
  for (int r = 0; r < 8; ++r) { mrow[r] = -INFINITY; lrow[r] = 0.0f; }

  const bf16* kp = qkv + (size_t)(b * SEQ) * C3 + N_EMBD + h * HEAD_DIM;
  const bf16* vp = qkv + (size_t)(b * SEQ) * C3 + 2 * N_EMBD + h * HEAD_DIM;
  const int ktmax = (qbase + 127) >> 6;

  for (int kt = 0; kt <= ktmax; ++kt) {
#pragma unroll
    for (int t = 0; t < 2; ++t) {              // K tile 64x64 (async)
      const int idx = tid + t * 256;
      const int row = idx >> 3, c = (idx & 7) * 8;
      cp16(&sK[row][c], kp + (size_t)(kt * 64 + row) * C3 + c);
    }
#pragma unroll
    for (int t = 0; t < 16; ++t) {             // V tile 64x64, transposed
      const int idx = tid + t * 256;
      const int s = idx >> 6, d = idx & 63;
      sVt[d][s] = vp[(size_t)(kt * 64 + s) * C3 + d];
    }
    wait_async();
    __syncthreads();

    v8f sacc[4];
#pragma unroll
    for (int nt = 0; nt < 4; ++nt) sacc[nt] = zero8();
#pragma unroll
    for (int ks = 0; ks < 2; ++ks) {
#pragma unroll
      for (int nt = 0; nt < 4; ++nt) {
        const int n = nt * 16 + l16;
        const v16bf kf = pack16(*(const v8bf*)&sK[n][ks * 32 + hlf * 8],
                                *(const v8bf*)&sK[n][ks * 32 + 16 + hlf * 8]);
        sacc[nt] = wmma_bf16f32(qf[ks], kf, sacc[nt]);
      }
    }

    // online softmax; each half-wave owns 8 rows of the 16-row strip
#pragma unroll
    for (int r = 0; r < 8; ++r) {
      const int q = qbase + wid * 16 + r + hlf * 8;
      float rmax = -INFINITY;
#pragma unroll
      for (int nt = 0; nt < 4; ++nt) {
        const int s = kt * 64 + nt * 16 + l16;
        float v = sacc[nt][r] * 0.125f;        // 1/sqrt(64)
        if (s > q) v = -INFINITY;
        sacc[nt][r] = v;
        rmax = fmaxf(rmax, v);
      }
#pragma unroll
      for (int off = 1; off < 16; off <<= 1)
        rmax = fmaxf(rmax, __shfl_xor(rmax, off, 32));
      const float mnew = fmaxf(mrow[r], rmax);
      const float alpha = (mnew == -INFINITY) ? 1.0f : __expf(mrow[r] - mnew);
      float rsum = 0.0f;
#pragma unroll
      for (int nt = 0; nt < 4; ++nt) {
        const float p = (mnew == -INFINITY) ? 0.0f : __expf(sacc[nt][r] - mnew);
        sP[wid * 16 + r + hlf * 8][nt * 16 + l16] = (bf16)p;
        rsum += p;
      }
#pragma unroll
      for (int off = 1; off < 16; off <<= 1) rsum += __shfl_xor(rsum, off, 32);
      lrow[r] = lrow[r] * alpha + rsum;
      mrow[r] = mnew;
#pragma unroll
      for (int dt = 0; dt < 4; ++dt) o[dt][r] *= alpha;
    }
    __syncthreads();

    // O += P V
#pragma unroll
    for (int ks = 0; ks < 2; ++ks) {
      const int r = wid * 16 + l16;
      const v16bf pf = pack16(*(const v8bf*)&sP[r][ks * 32 + hlf * 8],
                              *(const v8bf*)&sP[r][ks * 32 + 16 + hlf * 8]);
#pragma unroll
      for (int dt = 0; dt < 4; ++dt) {
        const int n = dt * 16 + l16;
        const v16bf vf = pack16(*(const v8bf*)&sVt[n][ks * 32 + hlf * 8],
                                *(const v8bf*)&sVt[n][ks * 32 + 16 + hlf * 8]);
        o[dt] = wmma_bf16f32(pf, vf, o[dt]);
      }
    }
    __syncthreads();
  }

  bf16* yp = y + (size_t)(b * SEQ + qbase) * N_EMBD + h * HEAD_DIM;
#pragma unroll
  for (int dt = 0; dt < 4; ++dt) {
#pragma unroll
    for (int r = 0; r < 8; ++r) {
      const int row = wid * 16 + r + hlf * 8;
      yp[(size_t)row * N_EMBD + dt * 16 + l16] = (bf16)(o[dt][r] / lrow[r]);
    }
  }
}

// ---------------------------------------------------------------------------
extern "C" void kernel_launch(void* const* d_in, const int* in_sizes, int n_in,
                              void* d_out, int out_size, void* d_ws,
                              size_t ws_size, hipStream_t stream) {
  const float* x       = (const float*)d_in[0];
  const float* ln1_w   = (const float*)d_in[1];
  const float* ln1_b   = (const float*)d_in[2];
  const float* ln2_w   = (const float*)d_in[3];
  const float* ln2_b   = (const float*)d_in[4];
  const float* w_attn  = (const float*)d_in[5];
  const float* b_attn  = (const float*)d_in[6];
  const float* w_attnp = (const float*)d_in[7];
  const float* b_attnp = (const float*)d_in[8];
  const float* w_fc    = (const float*)d_in[9];
  const float* b_fc    = (const float*)d_in[10];
  const float* w_proj  = (const float*)d_in[11];
  const float* b_proj  = (const float*)d_in[12];
  float* out = (float*)d_out;
  (void)in_sizes; (void)n_in; (void)out_size; (void)ws_size;

  char* p = (char*)d_ws;
  auto take = [&p](size_t bytes) -> char* {
    char* r = p;
    p += (bytes + 255) & ~(size_t)255;
    return r;
  };
  bf16*  wtA  = (bf16*)take((size_t)3072 * 1024 * 2);
  bf16*  wtAP = (bf16*)take((size_t)1024 * 1024 * 2);
  bf16*  wtFC = (bf16*)take((size_t)4096 * 1024 * 2);
  bf16*  wtPR = (bf16*)take((size_t)1024 * 4096 * 2);
  bf16*  ln1o = (bf16*)take((size_t)NTOK * 1024 * 2);
  bf16*  qkvb = (bf16*)take((size_t)NTOK * 3072 * 2);
  bf16*  yb   = (bf16*)take((size_t)NTOK * 1024 * 2);
  float* x1   = (float*)take((size_t)NTOK * 1024 * 4);
  bf16*  ln2o = (bf16*)take((size_t)NTOK * 1024 * 2);
  bf16*  hb   = (bf16*)take((size_t)NTOK * 4096 * 2);

  const dim3 blk(256);
  transpose_cvt_kernel<<<dim3(3072 / 32, 1024 / 32), blk, 0, stream>>>(w_attn, wtA, 1024, 3072);
  transpose_cvt_kernel<<<dim3(1024 / 32, 1024 / 32), blk, 0, stream>>>(w_attnp, wtAP, 1024, 1024);
  transpose_cvt_kernel<<<dim3(4096 / 32, 1024 / 32), blk, 0, stream>>>(w_fc, wtFC, 1024, 4096);
  transpose_cvt_kernel<<<dim3(1024 / 32, 4096 / 32), blk, 0, stream>>>(w_proj, wtPR, 4096, 1024);

  layernorm_kernel<<<NTOK, blk, 0, stream>>>(x, ln1_w, ln1_b, ln1o);
  gemm_bf16_kernel<0><<<dim3(3072 / 128, NTOK / 128), blk, 0, stream>>>(
      ln1o, wtA, b_attn, nullptr, qkvb, nullptr, NTOK, 3072, 1024);
  attn_kernel<<<dim3(SEQ / 128, N_HEAD, BATCH), blk, 0, stream>>>(qkvb, yb);
  gemm_bf16_kernel<1><<<dim3(1024 / 128, NTOK / 128), blk, 0, stream>>>(
      yb, wtAP, b_attnp, x, nullptr, x1, NTOK, 1024, 1024);
  layernorm_kernel<<<NTOK, blk, 0, stream>>>(x1, ln2_w, ln2_b, ln2o);
  gemm_bf16_kernel<2><<<dim3(4096 / 128, NTOK / 128), blk, 0, stream>>>(
      ln2o, wtFC, b_fc, nullptr, hb, nullptr, NTOK, 4096, 1024);
  gemm_bf16_kernel<1><<<dim3(1024 / 128, NTOK / 128), blk, 0, stream>>>(
      hb, wtPR, b_proj, x1, nullptr, out, NTOK, 1024, 4096);
}
